// WeightOnlyInt8Linear_39006892982811
// MI455X (gfx1250) — compile-verified, matrix operand order checked by reading, not run on hardware
//
#include <hip/hip_runtime.h>

// Problem dims (fixed by reference)
#define DIN   4096
#define DOUT  11008
#define MTOT  8192   // B*S = 4*2048

// LDS row pitch in halves (64 K-elems + 8 pad to break bank conflicts)
#define LDA 72

typedef _Float16 v16h __attribute__((ext_vector_type(16)));
typedef _Float16 v8h  __attribute__((ext_vector_type(8)));
typedef _Float16 v4h  __attribute__((ext_vector_type(4)));
typedef float    v8f  __attribute__((ext_vector_type(8)));
typedef float    fx4  __attribute__((ext_vector_type(4)));
typedef int      ix4  __attribute__((ext_vector_type(4)));

__global__ __launch_bounds__(256)
void int8lin_wmma_f16(const float* __restrict__ A,        // [MTOT, DIN] fp32
                      const signed char* __restrict__ W,  // [DOUT, DIN] int8
                      const float* __restrict__ scales,   // [DOUT] fp32
                      float* __restrict__ out)            // [MTOT, DOUT] fp32
{
    __shared__ _Float16 As[128 * LDA];   // 18 KB
    __shared__ _Float16 Bs[128 * LDA];   // 18 KB

    const int tid   = threadIdx.x;
    const int lane  = tid & 31;
    const int wave  = tid >> 5;          // 0..7
    const int waveM = wave & 1;          // 2 wave-rows  (M)
    const int waveN = wave >> 1;         // 4 wave-cols  (N)
    const int bN    = blockIdx.x;        // DOUT/128 = 86
    const int bM    = blockIdx.y;        // MTOT/128 = 64

    const int lh = lane >> 4;            // half-wave select (0/1)
    const int ln = lane & 15;

    // fp32 accumulators: 4 (M) x 2 (N) tiles of 16x16
    v8f acc[4][2];
    #pragma unroll
    for (int i = 0; i < 4; ++i)
        #pragma unroll
        for (int j = 0; j < 2; ++j)
            acc[i][j] = (v8f)0.0f;

    // per-out-channel scales for this lane's two columns
    float sc[2];
    #pragma unroll
    for (int j = 0; j < 2; ++j)
        sc[j] = scales[bN * 128 + waveN * 32 + j * 16 + ln];

    const size_t aBase = (size_t)bM * 128 * DIN;
    const size_t wBase = (size_t)bN * 128 * DIN;

    for (int k0 = 0; k0 < DIN; k0 += 64) {
        // ---- global loads for this K-slab ----
        // A tile: 128 rows x 64 K fp32 = 2048 float4 -> 8 per thread
        fx4 areg[8];
        #pragma unroll
        for (int i = 0; i < 8; ++i) {
            int q   = tid + i * 256;
            int row = q >> 4;            // /16 float4 per row
            int kc  = q & 15;
            areg[i] = *(const fx4*)(A + aBase + (size_t)row * DIN + k0 + kc * 4);
        }
        // W tile: 128 rows x 64 K int8 = 512 x 16B -> 2 per thread
        ix4 breg[2];
        #pragma unroll
        for (int i = 0; i < 2; ++i) {
            int q   = tid + i * 256;
            int row = q >> 2;            // /4 16B-chunks per row
            int kc  = q & 3;
            breg[i] = *(const ix4*)(W + wBase + (size_t)row * DIN + k0 + kc * 16);
        }

        // prefetch next K-slab into caches (gfx1250 global_prefetch_b8)
        if (k0 + 64 < DIN) {
            __builtin_prefetch(A + aBase + (size_t)(tid >> 4) * DIN + (k0 + 64) + (tid & 15) * 4, 0, 3);
            __builtin_prefetch(W + wBase + (size_t)(tid >> 2) * DIN + (k0 + 64) + (tid & 3) * 16, 0, 3);
        }

        __syncthreads();   // previous slab's compute done before overwrite

        // ---- convert once and stage to LDS as f16 ----
        #pragma unroll
        for (int i = 0; i < 8; ++i) {
            int q = tid + i * 256;
            int row = q >> 4, kc = q & 15;
            v4h h;
            h[0] = (_Float16)areg[i][0];
            h[1] = (_Float16)areg[i][1];
            h[2] = (_Float16)areg[i][2];
            h[3] = (_Float16)areg[i][3];
            *(v4h*)&As[row * LDA + kc * 4] = h;
        }
        #pragma unroll
        for (int i = 0; i < 2; ++i) {
            int q = tid + i * 256;
            int row = q >> 2, kc = q & 3;
            const signed char* bb = (const signed char*)&breg[i];
            v8h h0, h1;
            #pragma unroll
            for (int e = 0; e < 8; ++e) {
                h0[e] = (_Float16)(float)bb[e];
                h1[e] = (_Float16)(float)bb[8 + e];
            }
            *(v8h*)&Bs[row * LDA + kc * 16]     = h0;
            *(v8h*)&Bs[row * LDA + kc * 16 + 8] = h1;
        }

        __syncthreads();

        // ---- 2 WMMA K-steps of 32 over the 64-wide slab ----
        #pragma unroll
        for (int ks = 0; ks < 64; ks += 32) {
            v16h af[4], bf[2];
            // A fragments: lanes 0-15 take K[0..7]+K[16..23], lanes 16-31 K[8..15]+K[24..31]
            #pragma unroll
            for (int i = 0; i < 4; ++i) {
                int row = waveM * 64 + i * 16 + ln;
                const _Float16* p = &As[row * LDA + ks + lh * 8];
                *(fx4*)&af[i]       = *(const fx4*)p;          // ds_load_b128
                *((fx4*)&af[i] + 1) = *(const fx4*)(p + 16);   // ds_load_b128
            }
            #pragma unroll
            for (int j = 0; j < 2; ++j) {
                int row = waveN * 32 + j * 16 + ln;
                const _Float16* p = &Bs[row * LDA + ks + lh * 8];
                *(fx4*)&bf[j]       = *(const fx4*)p;
                *((fx4*)&bf[j] + 1) = *(const fx4*)(p + 16);
            }
            #pragma unroll
            for (int i = 0; i < 4; ++i)
                #pragma unroll
                for (int j = 0; j < 2; ++j)
                    acc[i][j] = __builtin_amdgcn_wmma_f32_16x16x32_f16(
                        /*neg_a=*/false, af[i], /*neg_b=*/false, bf[j],
                        /*c_mod=*/(short)0, acc[i][j],
                        /*reuse_a=*/false, /*reuse_b=*/false);
        }
        __syncthreads();
    }

    // ---- epilogue: apply per-channel scale, store fp32 ----
    // C/D layout: VGPR r, lanes 0-15 -> M=r, N=lane ; lanes 16-31 -> M=r+8, N=lane-16
    #pragma unroll
    for (int i = 0; i < 4; ++i) {
        int rowBase = bM * 128 + waveM * 64 + i * 16 + lh * 8;
        #pragma unroll
        for (int j = 0; j < 2; ++j) {
            int col = bN * 128 + waveN * 32 + j * 16 + ln;
            float s = sc[j];
            #pragma unroll
            for (int r = 0; r < 8; ++r)
                out[(size_t)(rowBase + r) * DOUT + col] = acc[i][j][r] * s;
        }
    }
}

extern "C" void kernel_launch(void* const* d_in, const int* in_sizes, int n_in,
                              void* d_out, int out_size, void* d_ws, size_t ws_size,
                              hipStream_t stream) {
    const float*       A  = (const float*)d_in[0];        // input  [4,2048,4096] fp32
    const signed char* W  = (const signed char*)d_in[1];  // weight [11008,4096] int8
    const float*       Sc = (const float*)d_in[2];        // scales [11008] fp32
    float*             O  = (float*)d_out;                // out    [4,2048,11008] fp32

    dim3 grid(DOUT / 128, MTOT / 128);  // (86, 64)
    int8lin_wmma_f16<<<grid, 256, 0, stream>>>(A, W, Sc, O);
}
